// CausalSelfAttention_48438641164330
// MI455X (gfx1250) — matrix operand order repaired and checked by reading, not compiled, not run on hardware
//
#include <hip/hip_runtime.h>
#include <hip/hip_bf16.h>
#include <math.h>

typedef __attribute__((ext_vector_type(16))) __bf16 v16bf;
typedef __attribute__((ext_vector_type(8)))  __bf16 v8bf;
typedef __attribute__((ext_vector_type(8)))  float  v8f;
typedef __attribute__((ext_vector_type(4)))  unsigned int u32x4;
typedef __attribute__((ext_vector_type(4)))  int i32x4;
typedef __attribute__((ext_vector_type(8)))  int i32x8;

#define N_EMBD 1024
#define N_HEAD 16
#define D_K    64
#define B_SZ   2
#define T_SEQ  2048
#define M_ROWS (B_SZ * T_SEQ)   // 4096

#if __has_builtin(__builtin_amdgcn_tensor_load_to_lds) && \
    __has_builtin(__builtin_amdgcn_s_wait_tensorcnt)
#define USE_TDM 1
#else
#define USE_TDM 0
#endif

union V16U { v16bf v; v8bf h[2]; unsigned short s[16]; };

__device__ __forceinline__ unsigned short f2bf(float f) {
    union { float f; unsigned int u; } c; c.f = f;
    unsigned int u = c.u;
    u += 0x7FFFu + ((u >> 16) & 1u);          // round-to-nearest-even
    return (unsigned short)(u >> 16);
}

__device__ __forceinline__ void ld_a(V16U& a, const unsigned short* p) {
    a.h[0] = *(const v8bf*)p;  a.h[1] = *(const v8bf*)(p + 8);
}
__device__ __forceinline__ void ld_b(V16U& b, const unsigned short* p) {
    b.h[0] = *(const v8bf*)p;  b.h[1] = *(const v8bf*)(p + 8);
}
__device__ __forceinline__ v8f wmma_bf16(const V16U& a, const V16U& b, v8f c) {
    return __builtin_amdgcn_wmma_f32_16x16x32_bf16(false, a.v, false, b.v,
                                                   (short)0, c, false, false);
}

#if USE_TDM
// 2-D TDM load: tile_d1 rows x tile_d0 elements (2-byte), row stride stride0
// elements, from global gptr into LDS byte-offset lds_off. D# layout per
// CDNA5 ISA (group0 128b, group1 256b).
__device__ __forceinline__ void tdm_load_2d(unsigned lds_off,
                                            const unsigned short* gptr,
                                            unsigned tile_d0, unsigned tile_d1,
                                            unsigned tensor_d0, unsigned tensor_d1,
                                            unsigned stride0) {
    unsigned long long ga = (unsigned long long)(uintptr_t)gptr;
    u32x4 g0;
    g0[0] = 1u;                                             // count=1 (user D#)
    g0[1] = lds_off;                                        // lds_addr (bytes)
    g0[2] = (unsigned)ga;                                   // global_addr[31:0]
    g0[3] = ((unsigned)(ga >> 32) & 0x01FFFFFFu) | 0x80000000u; // [56:32] | type=2
    i32x8 g1;
    g1[0] = 0x00010000;                                     // data_size=1 -> 2B
    g1[1] = (int)((tensor_d0 & 0xFFFFu) << 16);             // tdim0[15:0]
    g1[2] = (int)(((tensor_d0 >> 16) & 0xFFFFu) |
                  ((tensor_d1 & 0xFFFFu) << 16));           // tdim0[31:16], tdim1[15:0]
    g1[3] = (int)(((tensor_d1 >> 16) & 0xFFFFu) |
                  ((tile_d0 & 0xFFFFu) << 16));             // tdim1[31:16], tile_d0
    g1[4] = (int)(tile_d1 & 0xFFFFu);                       // tile_d1, tile_d2=0
    g1[5] = (int)stride0;                                   // dim0 stride [31:0]
    g1[6] = 0;                                              // stride0 hi, stride1 lo
    g1[7] = 0;
    i32x4 z4 = {};
#if defined(__clang_major__) && __clang_major__ >= 23
    i32x8 z8 = {};
    __builtin_amdgcn_tensor_load_to_lds(g0, g1, z4, z4, z8, 0);
#else
    __builtin_amdgcn_tensor_load_to_lds(g0, g1, z4, z4, 0);
#endif
}
#endif

// ---------------------------------------------------------------- convert ---
__global__ void cvt_f32_bf16(const float* __restrict__ src,
                             unsigned short* __restrict__ dst, int n) {
    int i = blockIdx.x * blockDim.x + threadIdx.x;
    if (i < n) dst[i] = f2bf(src[i]);
}

// ----------------------------------------------------- 32x64 GEMM core ------
__device__ __forceinline__ void gemm32x64(const unsigned short* __restrict__ ab,
                                          const unsigned short* __restrict__ wb,
                                          int lo, int hi, v8f acc[2][4]) {
    const unsigned short* ap[2];
    ap[0] = ab + lo * N_EMBD + hi * 16;
    ap[1] = ab + (16 + lo) * N_EMBD + hi * 16;
    const unsigned short* bp[4];
#pragma unroll
    for (int t = 0; t < 4; ++t) bp[t] = wb + (t * 16 + lo) * N_EMBD + hi * 16;

    V16U A[2][2], Bv[2][4];
#pragma unroll
    for (int u = 0; u < 2; ++u) ld_a(A[0][u], ap[u]);
#pragma unroll
    for (int t = 0; t < 4; ++t) ld_b(Bv[0][t], bp[t]);

    for (int kk = 0; kk < N_EMBD; kk += 64) {
#pragma unroll
        for (int u = 0; u < 2; ++u) ld_a(A[1][u], ap[u] + kk + 32);
#pragma unroll
        for (int t = 0; t < 4; ++t) ld_b(Bv[1][t], bp[t] + kk + 32);
#pragma unroll
        for (int t = 0; t < 4; ++t)
#pragma unroll
            for (int u = 0; u < 2; ++u)
                acc[u][t] = wmma_bf16(A[0][u], Bv[0][t], acc[u][t]);
        if (kk + 64 < N_EMBD) {
#pragma unroll
            for (int u = 0; u < 2; ++u) ld_a(A[0][u], ap[u] + kk + 64);
#pragma unroll
            for (int t = 0; t < 4; ++t) ld_b(Bv[0][t], bp[t] + kk + 64);
        }
#pragma unroll
        for (int t = 0; t < 4; ++t)
#pragma unroll
            for (int u = 0; u < 2; ++u)
                acc[u][t] = wmma_bf16(A[1][u], Bv[1][t], acc[u][t]);
    }
}

// ------------------------------------------------------------- QKV GEMM -----
__global__ __launch_bounds__(128)
void qkv_gemm(const unsigned short* __restrict__ xb,
              const unsigned short* __restrict__ wq,
              const unsigned short* __restrict__ wk,
              const unsigned short* __restrict__ wv,
              const float* __restrict__ bq,
              const float* __restrict__ bk,
              const float* __restrict__ bv,
              unsigned short* __restrict__ qb,
              unsigned short* __restrict__ kb,
              unsigned short* __restrict__ vtb) {
    const int L  = threadIdx.x & 31;
    const int wid = threadIdx.x >> 5;
    const int lo = L & 15, hi = L >> 4;
    const int mt = blockIdx.x * 4 + wid;
    const int nt = blockIdx.y;
    const int which = blockIdx.z;
    const unsigned short* wmat = (which == 0) ? wq : (which == 1) ? wk : wv;
    const float* bias          = (which == 0) ? bq : (which == 1) ? bk : bv;

    v8f acc[2][4] = {};
    gemm32x64(xb + (mt * 32) * N_EMBD, wmat + (nt * 64) * N_EMBD, lo, hi, acc);

#pragma unroll
    for (int t = 0; t < 4; ++t) {
        const int j  = nt * 64 + t * 16 + lo;
        const float bj = bias[j];
        const int hd = j >> 6, d = j & 63;
#pragma unroll
        for (int u = 0; u < 2; ++u)
#pragma unroll
            for (int r = 0; r < 8; ++r) {
                const int i    = mt * 32 + u * 16 + r + 8 * hi;
                const int bidx = i >> 11, tpos = i & (T_SEQ - 1);
                const int bh   = bidx * N_HEAD + hd;
                const float vo = acc[u][t][r] + bj;
                if (which == 0)
                    qb[(bh * T_SEQ + tpos) * D_K + d] = f2bf(vo * 0.125f);
                else if (which == 1)
                    kb[(bh * T_SEQ + tpos) * D_K + d] = f2bf(vo);
                else
                    vtb[(bh * D_K + d) * T_SEQ + tpos] = f2bf(vo);
            }
    }
}

// ---------------------------------------------------------- attention -------
// One wave per 16-query block; flash online softmax over key blocks of 32.
// K/V blocks streamed into double-buffered LDS by the Tensor Data Mover,
// synchronized with TENSORcnt; WMMAs consume the previous block meanwhile.
__global__ __launch_bounds__(32)
void attn(const unsigned short* __restrict__ qbuf,
          const unsigned short* __restrict__ kbuf,
          const unsigned short* __restrict__ vtbuf,
          unsigned short* __restrict__ ybuf) {
    __shared__ unsigned short pbuf[16 * 32];
#if USE_TDM
    __shared__ unsigned short kls[2][32 * 64];   // K block: 32 keys x 64 dims
    __shared__ unsigned short vls[2][64 * 32];   // V block: 64 dims x 32 keys
#endif
    const int L  = threadIdx.x;
    const int lo = L & 15, hi = L >> 4;
    const int qt = blockIdx.x;
    const int bh = blockIdx.y;
    const int qbase = qt * 16;

    const unsigned short* qp = qbuf + (bh * T_SEQ + qbase) * D_K;
    const unsigned short* kp = kbuf + bh * T_SEQ * D_K;
    const unsigned short* vp = vtbuf + bh * D_K * T_SEQ;

    V16U qa[2];
#pragma unroll
    for (int c = 0; c < 2; ++c) {
        const unsigned short* p = qp + lo * D_K + c * 32 + hi * 16;
        qa[c].h[0] = *(const v8bf*)p;
        qa[c].h[1] = *(const v8bf*)(p + 8);
    }

    v8f o[4] = {};
    float mrow[8], lrow[8];
#pragma unroll
    for (int r = 0; r < 8; ++r) { mrow[r] = -1e30f; lrow[r] = 0.f; }

    const int kend = qbase + 16;
#if USE_TDM
    // prologue: stream first K/V block into LDS buffer 0
    tdm_load_2d((unsigned)(uintptr_t)&kls[0][0], kp, D_K, 32, D_K, T_SEQ, D_K);
    tdm_load_2d((unsigned)(uintptr_t)&vls[0][0], vp, 32, D_K, T_SEQ, D_K, T_SEQ);
#endif
    for (int kbk = 0; kbk < kend; kbk += 32) {
        V16U kb0[2], kb1[2], vb[4];
#if USE_TDM
        const int buf = (kbk >> 5) & 1;
        if (kbk + 32 < kend) {    // stream next block while consuming this one
            tdm_load_2d((unsigned)(uintptr_t)&kls[buf ^ 1][0],
                        kp + (kbk + 32) * D_K, D_K, 32, D_K, T_SEQ, D_K);
            tdm_load_2d((unsigned)(uintptr_t)&vls[buf ^ 1][0],
                        vp + (kbk + 32), 32, D_K, T_SEQ, D_K, T_SEQ);
            __builtin_amdgcn_s_wait_tensorcnt(2);
        } else {
            __builtin_amdgcn_s_wait_tensorcnt(0);
        }
#pragma unroll
        for (int c = 0; c < 2; ++c) {
            ld_b(kb0[c], &kls[buf][lo * D_K        + c * 32 + hi * 16]);
            ld_b(kb1[c], &kls[buf][(16 + lo) * D_K + c * 32 + hi * 16]);
        }
#pragma unroll
        for (int t = 0; t < 4; ++t)
            ld_b(vb[t], &vls[buf][(t * 16 + lo) * 32 + hi * 16]);
#else
#pragma unroll
        for (int c = 0; c < 2; ++c) {
            ld_b(kb0[c], kp + (kbk + lo) * D_K + c * 32 + hi * 16);
            ld_b(kb1[c], kp + (kbk + 16 + lo) * D_K + c * 32 + hi * 16);
        }
        if (kbk + 32 < kend) {
            __builtin_prefetch(kp + (kbk + 32 + L) * D_K, 0, 3);
            __builtin_prefetch(vp + ((L & 3) * 16 + (L >> 2)) * T_SEQ + kbk + 32, 0, 3);
        }
#pragma unroll
        for (int t = 0; t < 4; ++t)
            ld_b(vb[t], vp + (t * 16 + lo) * T_SEQ + kbk + hi * 16);
#endif
        // ---- S = Q * K^T
        v8f s0 = {}, s1 = {};
        s0 = wmma_bf16(qa[0], kb0[0], s0);
        s0 = wmma_bf16(qa[1], kb0[1], s0);
        s1 = wmma_bf16(qa[0], kb1[0], s1);
        s1 = wmma_bf16(qa[1], kb1[1], s1);
        // ---- causal mask + online softmax
#pragma unroll
        for (int r = 0; r < 8; ++r) {
            const int row = qbase + r + 8 * hi;
            float x0 = (kbk + lo      <= row) ? s0[r] : -1e30f;
            float x1 = (kbk + 16 + lo <= row) ? s1[r] : -1e30f;
            float mx = fmaxf(x0, x1);
            mx = fmaxf(mx, __shfl_xor(mx, 1, 32));
            mx = fmaxf(mx, __shfl_xor(mx, 2, 32));
            mx = fmaxf(mx, __shfl_xor(mx, 4, 32));
            mx = fmaxf(mx, __shfl_xor(mx, 8, 32));
            const float mnew  = fmaxf(mrow[r], mx);
            const float alpha = __expf(mrow[r] - mnew);
            const float e0 = __expf(x0 - mnew);
            const float e1 = __expf(x1 - mnew);
            float rs = e0 + e1;
            rs += __shfl_xor(rs, 1, 32);
            rs += __shfl_xor(rs, 2, 32);
            rs += __shfl_xor(rs, 4, 32);
            rs += __shfl_xor(rs, 8, 32);
            lrow[r] = lrow[r] * alpha + rs;
            mrow[r] = mnew;
#pragma unroll
            for (int t = 0; t < 4; ++t) o[t][r] *= alpha;
            pbuf[(r + 8 * hi) * 32 + lo]      = f2bf(e0);
            pbuf[(r + 8 * hi) * 32 + lo + 16] = f2bf(e1);
        }
        __syncthreads();                      // C-layout -> A-layout via LDS
        V16U pa;
        const unsigned short* pr = &pbuf[lo * 32 + hi * 16];
        pa.h[0] = *(const v8bf*)pr;
        pa.h[1] = *(const v8bf*)(pr + 8);
        __syncthreads();
        // ---- O += P * V
#pragma unroll
        for (int t = 0; t < 4; ++t)
            o[t] = wmma_bf16(pa, vb[t], o[t]);
    }
    // ---- normalize and store y
    const int b = bh / N_HEAD, hd = bh % N_HEAD;
#pragma unroll
    for (int r = 0; r < 8; ++r) {
        const int row = qbase + r + 8 * hi;
        const float inv = 1.0f / lrow[r];
#pragma unroll
        for (int t = 0; t < 4; ++t)
            ybuf[(b * T_SEQ + row) * N_EMBD + hd * D_K + t * 16 + lo] =
                f2bf(o[t][r] * inv);
    }
}

// ------------------------------------------------------------ out GEMM ------
__global__ __launch_bounds__(128)
void out_gemm(const unsigned short* __restrict__ yb,
              const unsigned short* __restrict__ wo,
              const float* __restrict__ bo,
              float* __restrict__ out) {
    const int L  = threadIdx.x & 31;
    const int wid = threadIdx.x >> 5;
    const int lo = L & 15, hi = L >> 4;
    const int mt = blockIdx.x * 4 + wid;
    const int nt = blockIdx.y;

    v8f acc[2][4] = {};
    gemm32x64(yb + (mt * 32) * N_EMBD, wo + (nt * 64) * N_EMBD, lo, hi, acc);

#pragma unroll
    for (int t = 0; t < 4; ++t) {
        const int j = nt * 64 + t * 16 + lo;
        const float bj = bo[j];
#pragma unroll
        for (int u = 0; u < 2; ++u)
#pragma unroll
            for (int r = 0; r < 8; ++r) {
                const int i = mt * 32 + u * 16 + r + 8 * hi;
                out[i * N_EMBD + j] = acc[u][t][r] + bj;
            }
    }
}

// ------------------------------------------------------------- launcher -----
extern "C" void kernel_launch(void* const* d_in, const int* in_sizes, int n_in,
                              void* d_out, int out_size, void* d_ws, size_t ws_size,
                              hipStream_t stream) {
    const float* x  = (const float*)d_in[0];
    const float* Wq = (const float*)d_in[1];
    const float* bq = (const float*)d_in[2];
    const float* Wk = (const float*)d_in[3];
    const float* bk = (const float*)d_in[4];
    const float* Wv = (const float*)d_in[5];
    const float* bv = (const float*)d_in[6];
    const float* Wo = (const float*)d_in[7];
    const float* bo = (const float*)d_in[8];
    float* out = (float*)d_out;

    char* w = (char*)d_ws;
    const size_t SX = (size_t)M_ROWS * N_EMBD * 2;
    const size_t SW = (size_t)N_EMBD * N_EMBD * 2;
    unsigned short* xb  = (unsigned short*)(w);
    unsigned short* wqb = (unsigned short*)(w + SX);
    unsigned short* wkb = (unsigned short*)(w + SX + 1 * SW);
    unsigned short* wvb = (unsigned short*)(w + SX + 2 * SW);
    unsigned short* wob = (unsigned short*)(w + SX + 3 * SW);
    unsigned short* qb  = (unsigned short*)(w + SX + 4 * SW);
    unsigned short* kbf = (unsigned short*)(w + 2 * SX + 4 * SW);
    unsigned short* vtb = (unsigned short*)(w + 3 * SX + 4 * SW);
    unsigned short* yb  = (unsigned short*)(w + 4 * SX + 4 * SW);

    const int nact = M_ROWS * N_EMBD;
    const int nwts = N_EMBD * N_EMBD;
    cvt_f32_bf16<<<(nact + 255) / 256, 256, 0, stream>>>(x,  xb,  nact);
    cvt_f32_bf16<<<(nwts + 255) / 256, 256, 0, stream>>>(Wq, wqb, nwts);
    cvt_f32_bf16<<<(nwts + 255) / 256, 256, 0, stream>>>(Wk, wkb, nwts);
    cvt_f32_bf16<<<(nwts + 255) / 256, 256, 0, stream>>>(Wv, wvb, nwts);
    cvt_f32_bf16<<<(nwts + 255) / 256, 256, 0, stream>>>(Wo, wob, nwts);

    qkv_gemm<<<dim3(M_ROWS / 32 / 4, N_EMBD / 64, 3), 128, 0, stream>>>(
        xb, wqb, wkb, wvb, bq, bk, bv, qb, kbf, vtb);

    attn<<<dim3(T_SEQ / 16, B_SZ * N_HEAD), 32, 0, stream>>>(qb, kbf, vtb, yb);

    out_gemm<<<dim3(M_ROWS / 32 / 4, N_EMBD / 64), 128, 0, stream>>>(yb, wob, bo, out);
}
